// MSDeformAttention_43593918054475
// MI455X (gfx1250) — matrix operand order repaired and checked by reading, not compiled, not run on hardware
//
#include <hip/hip_runtime.h>
#include <hip/hip_bf16.h>

typedef __attribute__((ext_vector_type(16))) _Float16 v16h;
typedef __attribute__((ext_vector_type(8)))  _Float16 v8h;
typedef __attribute__((ext_vector_type(8)))  float    v8f;

#define BSZ   8
#define LQ    5440
#define MTOT  (BSZ * LQ)   // 43520 = 340 * 128
#define DDIM  256

// ---------------------------------------------------------------------------
// Elementwise f32 -> f16 conversion (grid-stride)
// ---------------------------------------------------------------------------
__global__ __launch_bounds__(256) void k_f32_to_f16(const float* __restrict__ in,
                                                    _Float16* __restrict__ out, int n) {
  int i = blockIdx.x * blockDim.x + threadIdx.x;
  int stride = gridDim.x * blockDim.x;
  for (; i < n; i += stride) out[i] = (_Float16)in[i];
}

// W [K][N] f32 -> WT [N][K] f16 (tiny weight matrices)
__global__ __launch_bounds__(256) void k_transpose_f16(const float* __restrict__ W,
                                                       _Float16* __restrict__ WT,
                                                       int K, int N) {
  int i = blockIdx.x * blockDim.x + threadIdx.x;
  if (i < K * N) {
    int k = i / N;
    int n = i - k * N;
    WT[n * K + k] = (_Float16)W[i];
  }
}

// ---------------------------------------------------------------------------
// GEMM: C[M,N] = A[M,K] (f16 row-major) * W + bias, with BT = W^T [N,K] f16.
// Output f16 (OUTF16=true) or f32. Block tile 128x128, K-step 32, 8 waves;
// wave (wm 0..3, wn 0..1) owns 2(M) x 4(N) wmma 16x16 tiles.
// Software-pipelined: next K-tile global loads issue before the WMMA work.
// ---------------------------------------------------------------------------
#define BM 128
#define BN 128
#define BK 32
#define LDT 40  // padded LDS row stride in halfs (80B, 16B-aligned, conflict-free)

template <bool OUTF16>
__global__ __launch_bounds__(256) void k_gemm_f16(
    const _Float16* __restrict__ A,    // [M,K]
    const _Float16* __restrict__ BT,   // [N,K]
    const float*    __restrict__ bias, // [N] (never null)
    void* __restrict__ Cout,
    int M, int N, int K) {
  __shared__ __align__(16) _Float16 As[BM * LDT];
  __shared__ __align__(16) _Float16 Bs[BN * LDT];

  const int tid  = threadIdx.x;
  const int lane = tid & 31;
  const int wave = tid >> 5;
  const int wm   = wave & 3;   // M offset wm*32
  const int wn   = wave >> 2;  // N offset wn*64
  const int half = lane >> 4;  // 0 or 1
  const int l16  = lane & 15;

  const int row0 = blockIdx.x * BM;
  const int col0 = blockIdx.y * BN;

  const int ldr = tid >> 1;        // 0..127 tile row
  const int ldp = (tid & 1) * 16;  // 0 or 16 halfs

  const _Float16* ag = A  + (size_t)(row0 + ldr) * K + ldp;
  const _Float16* bg = BT + (size_t)(col0 + ldr) * K + ldp;
  uint4* adst = (uint4*)(As + ldr * LDT + ldp);
  uint4* bdst = (uint4*)(Bs + ldr * LDT + ldp);

  v8f acc[2][4];
  const v8f vzero = {0.f, 0.f, 0.f, 0.f, 0.f, 0.f, 0.f, 0.f};
#pragma unroll
  for (int i = 0; i < 2; ++i)
#pragma unroll
    for (int j = 0; j < 4; ++j) acc[i][j] = vzero;

  // prologue: first tile into registers
  uint4 ra0 = ((const uint4*)ag)[0];
  uint4 ra1 = ((const uint4*)ag)[1];
  uint4 rb0 = ((const uint4*)bg)[0];
  uint4 rb1 = ((const uint4*)bg)[1];

  for (int k0 = 0; k0 < K; k0 += BK) {
    adst[0] = ra0;
    adst[1] = ra1;
    bdst[0] = rb0;
    bdst[1] = rb1;
    __syncthreads();

    // prefetch next K-tile while this one is consumed by WMMA
    if (k0 + BK < K) {
      ag += BK;
      bg += BK;
      ra0 = ((const uint4*)ag)[0];
      ra1 = ((const uint4*)ag)[1];
      rb0 = ((const uint4*)bg)[0];
      rb1 = ((const uint4*)bg)[1];
    }

    // A fragments: lane holds row (l16); K = 16*half .. 16*half+15 contiguous
    v16h af[2];
#pragma unroll
    for (int mi = 0; mi < 2; ++mi) {
      const _Float16* rp = As + (wm * 32 + mi * 16 + l16) * LDT;
      v8h c0 = *(const v8h*)(rp + 16 * half);
      v8h c1 = *(const v8h*)(rp + 16 * half + 8);
      af[mi] = __builtin_shufflevector(c0, c1, 0, 1, 2, 3, 4, 5, 6, 7, 8, 9, 10,
                                       11, 12, 13, 14, 15);
    }
    // B fragments: lane holds col (l16); K = 16*half .. 16*half+15 contiguous
#pragma unroll
    for (int ni = 0; ni < 4; ++ni) {
      const _Float16* cp = Bs + (wn * 64 + ni * 16 + l16) * LDT + 16 * half;
      v8h c0 = *(const v8h*)(cp);
      v8h c1 = *(const v8h*)(cp + 8);
      v16h bf = __builtin_shufflevector(c0, c1, 0, 1, 2, 3, 4, 5, 6, 7, 8, 9, 10,
                                        11, 12, 13, 14, 15);
#pragma unroll
      for (int mi = 0; mi < 2; ++mi) {
        acc[mi][ni] = __builtin_amdgcn_wmma_f32_16x16x32_f16(
            false, af[mi], false, bf, (short)0, acc[mi][ni], false, false);
      }
    }
    __syncthreads();
  }

  // Epilogue (branch-free): lane L, vgpr r -> M = r + 8*(L>>4), N = L&15
  _Float16* __restrict__ Ch = (_Float16*)Cout;
  float* __restrict__ Cf = (float*)Cout;
#pragma unroll
  for (int mi = 0; mi < 2; ++mi) {
#pragma unroll
    for (int ni = 0; ni < 4; ++ni) {
      const int col = col0 + wn * 64 + ni * 16 + l16;
      const float bv = bias[col];
#pragma unroll
      for (int r = 0; r < 8; ++r) {
        const int row = row0 + wm * 32 + mi * 16 + half * 8 + r;
        const float v = acc[mi][ni][r] + bv;
        if constexpr (OUTF16)
          Ch[(size_t)row * N + col] = (_Float16)v;
        else
          Cf[(size_t)row * N + col] = v;
      }
    }
  }
}

// ---------------------------------------------------------------------------
// Double softmax (in place): aw [43520, 128]; first over 128, then per-head
// over 16 (cols ordered (head, level, point)). One wave per query row.
// ---------------------------------------------------------------------------
__global__ __launch_bounds__(256) void k_softmax(float* __restrict__ aw) {
  __shared__ float buf[8][128];
  const int row  = threadIdx.x >> 5;
  const int lane = threadIdx.x & 31;
  const int q    = blockIdx.x * 8 + row;
  float* p = aw + (size_t)q * 128;

  float v[4];
#pragma unroll
  for (int i = 0; i < 4; ++i) v[i] = p[lane + 32 * i];
  float m = fmaxf(fmaxf(v[0], v[1]), fmaxf(v[2], v[3]));
#pragma unroll
  for (int o = 16; o > 0; o >>= 1) m = fmaxf(m, __shfl_xor(m, o, 32));
  float s = 0.f;
#pragma unroll
  for (int i = 0; i < 4; ++i) { v[i] = expf(v[i] - m); s += v[i]; }
#pragma unroll
  for (int o = 16; o > 0; o >>= 1) s += __shfl_xor(s, o, 32);
  const float inv = 1.0f / s;
#pragma unroll
  for (int i = 0; i < 4; ++i) buf[row][lane + 32 * i] = v[i] * inv;
  __syncthreads();

  if (lane < 8) {  // one lane per head: softmax over its 16 entries
    const float* hp = buf[row] + lane * 16;
    float mm = hp[0];
    for (int i = 1; i < 16; ++i) mm = fmaxf(mm, hp[i]);
    float t[16];
    float ss = 0.f;
    for (int i = 0; i < 16; ++i) { t[i] = expf(hp[i] - mm); ss += t[i]; }
    const float iv = 1.0f / ss;
    for (int i = 0; i < 16; ++i) p[lane * 16 + i] = t[i] * iv;
  }
}

// ---------------------------------------------------------------------------
// Bilinear sampling + attention-weighted sum.
// One wave per (b, q, head); lane = channel d (0..31).
// vh: [BS, LQ, 256] f16 (channel = head*32+d) — 22MB, L2-resident.
// ---------------------------------------------------------------------------
__global__ __launch_bounds__(256) void k_sample(
    const _Float16* __restrict__ vh,
    const float* __restrict__ off,   // [BS, LQ, 256]
    const float* __restrict__ aw,    // [BS, LQ, 128]
    const float* __restrict__ ref,   // [BS, LQ, 4, 2]
    _Float16* __restrict__ mid) {    // [BS, LQ, 256]
  const int sizes[4]  = {64, 32, 16, 8};
  const int starts[4] = {0, 4096, 5120, 5376};

  const int unit = blockIdx.x * 8 + (threadIdx.x >> 5);
  const int lane = threadIdx.x & 31;  // channel d
  const int h    = unit & 7;
  const int bq   = unit >> 3;  // b*LQ + q
  const int b    = bq / LQ;

  const float* offp = off + (size_t)bq * 256;
  const float* awp  = aw + (size_t)bq * 128 + h * 16;
  const float* refp = ref + (size_t)bq * 8;
  const size_t vbase = (size_t)b * LQ * 256 + h * 32 + lane;

  float acc = 0.f;
#pragma unroll
  for (int lvl = 0; lvl < 4; ++lvl) {
    const int   W_    = sizes[lvl];
    const int   pix0  = starts[lvl];
    const float rx    = refp[lvl * 2 + 0];
    const float ry    = refp[lvl * 2 + 1];
    const float invn  = 1.0f / (float)W_;        // norm (square levels)
    const float scale = 0.5f * (float)(W_ - 2);  // 0.5 * (max-1)
#pragma unroll
    for (int pt = 0; pt < 4; ++pt) {
      const int oc = ((h * 4 + lvl) * 4 + pt) * 2;
      const float cx = rx + offp[oc + 0] * invn;
      const float cy = ry + offp[oc + 1] * invn;
      const float x = (cx + 1.0f) * scale;
      const float y = (cy + 1.0f) * scale;
      const int x0 = (int)floorf(x);
      const int y0 = (int)floorf(y);
      const int x0c = min(max(x0, 0), W_ - 1);
      const int x1c = min(max(x0 + 1, 0), W_ - 1);
      const int y0c = min(max(y0, 0), W_ - 1);
      const int y1c = min(max(y0 + 1, 0), W_ - 1);
      const float wx0 = (float)x1c - x;
      const float wx1 = x - (float)x0c;
      const float wy0 = (float)y1c - y;
      const float wy1 = y - (float)y0c;
      const float Ia = (float)vh[vbase + (size_t)(pix0 + y0c * W_ + x0c) * 256];
      const float Ib = (float)vh[vbase + (size_t)(pix0 + y1c * W_ + x0c) * 256];
      const float Ic = (float)vh[vbase + (size_t)(pix0 + y0c * W_ + x1c) * 256];
      const float Id = (float)vh[vbase + (size_t)(pix0 + y1c * W_ + x1c) * 256];
      const float s = (wx0 * wy0) * Ia + (wx0 * wy1) * Ib + (wx1 * wy0) * Ic +
                      (wx1 * wy1) * Id;
      acc += awp[lvl * 4 + pt] * s;
    }
  }
  mid[(size_t)bq * 256 + h * 32 + lane] = (_Float16)acc;
}

// ---------------------------------------------------------------------------
// Host launcher
// ---------------------------------------------------------------------------
extern "C" void kernel_launch(void* const* d_in, const int* in_sizes, int n_in,
                              void* d_out, int out_size, void* d_ws, size_t ws_size,
                              hipStream_t stream) {
  (void)in_sizes; (void)n_in; (void)out_size; (void)ws_size;
  const float* query = (const float*)d_in[0];
  const float* value = (const float*)d_in[1];
  const float* refp  = (const float*)d_in[2];
  const float* Wv    = (const float*)d_in[3];
  const float* bv    = (const float*)d_in[4];
  const float* Woff  = (const float*)d_in[5];
  const float* boff  = (const float*)d_in[6];
  const float* Wa    = (const float*)d_in[7];
  const float* ba    = (const float*)d_in[8];
  const float* Wout  = (const float*)d_in[9];
  const float* bout  = (const float*)d_in[10];

  char* ws = (char*)d_ws;
  size_t o = 0;
  auto wsalloc = [&](size_t bytes) -> char* {
    char* p = ws + o;
    o += (bytes + 255) & ~(size_t)255;
    return p;
  };
  _Float16* q_h   = (_Float16*)wsalloc((size_t)MTOT * 256 * 2);
  _Float16* vin_h = (_Float16*)wsalloc((size_t)MTOT * 256 * 2);
  _Float16* v_h   = (_Float16*)wsalloc((size_t)MTOT * 256 * 2);
  _Float16* mid_h = (_Float16*)wsalloc((size_t)MTOT * 256 * 2);
  float*    off_f = (float*)wsalloc((size_t)MTOT * 256 * 4);
  float*    aw_f  = (float*)wsalloc((size_t)MTOT * 128 * 4);
  _Float16* WvT   = (_Float16*)wsalloc(256 * 256 * 2);
  _Float16* WoffT = (_Float16*)wsalloc(256 * 256 * 2);
  _Float16* WaT   = (_Float16*)wsalloc(256 * 128 * 2);
  _Float16* WoutT = (_Float16*)wsalloc(256 * 256 * 2);

  const int nelem = MTOT * 256;
  k_f32_to_f16<<<4096, 256, 0, stream>>>(query, q_h, nelem);
  k_f32_to_f16<<<4096, 256, 0, stream>>>(value, vin_h, nelem);
  k_transpose_f16<<<256, 256, 0, stream>>>(Wv, WvT, 256, 256);
  k_transpose_f16<<<256, 256, 0, stream>>>(Woff, WoffT, 256, 256);
  k_transpose_f16<<<128, 256, 0, stream>>>(Wa, WaT, 256, 128);
  k_transpose_f16<<<256, 256, 0, stream>>>(Wout, WoutT, 256, 256);

  dim3 g2(MTOT / 128, 2);
  dim3 g1(MTOT / 128, 1);
  // v = value @ Wv + bv  (f16 out, L2-resident for sampling)
  k_gemm_f16<true><<<g2, 256, 0, stream>>>(vin_h, WvT, bv, v_h, MTOT, 256, 256);
  // off = query @ Woff + boff (f32 out)
  k_gemm_f16<false><<<g2, 256, 0, stream>>>(q_h, WoffT, boff, off_f, MTOT, 256, 256);
  // aw_raw = query @ Wa + ba (f32 out)
  k_gemm_f16<false><<<g1, 256, 0, stream>>>(q_h, WaT, ba, aw_f, MTOT, 128, 256);
  // double softmax (in place)
  k_softmax<<<MTOT / 8, 256, 0, stream>>>(aw_f);
  // bilinear sampling + weighted sum -> mid (f16)
  k_sample<<<MTOT, 256, 0, stream>>>(v_h, off_f, aw_f, refp, mid_h);
  // out = mid @ Wout + bout (f32 final output)
  k_gemm_f16<false><<<g2, 256, 0, stream>>>(mid_h, WoutT, bout, (float*)d_out,
                                            MTOT, 256, 256);
}
